// LlamaAttention_23536420782098
// MI455X (gfx1250) — compile-verified, matrix operand order checked by reading, not run on hardware
//
#include <hip/hip_runtime.h>
#include <hip/hip_bf16.h>

typedef __attribute__((ext_vector_type(16))) __bf16 v16bf;
typedef __attribute__((ext_vector_type(8)))  __bf16 v8bf;
typedef __attribute__((ext_vector_type(8)))  float  v8f;

#define HID   4096
#define NH    32
#define NKVH  8
#define HD    128
#define RLOW  64
#define SC    32767          // cached tokens
#define STOT  32768          // SC + 1 new token
#define NGRP  8192           // STOT / 4
#define NTOK  404            // 100*4 + 4 window
#define RCHUNK 32            // split-K row chunks for GEMVs

// ---------------- K1a: fused QKV GEMV, split-K partials --------------------
// thread = 4 consecutive output columns (float4, coalesced b128);
// grid.y = row chunk of 128 rows; deterministic reduce in K1b.
__global__ void qkv_gemv_part_k(const float* __restrict__ h,
                                const float* __restrict__ Wq,
                                const float* __restrict__ Wk,
                                const float* __restrict__ Wv,
                                float* __restrict__ part) {
    int col = (blockIdx.x * blockDim.x + threadIdx.x) * 4;   // 0..6140
    int chunk = blockIdx.y;                                  // 0..31
    const float* W; int c, ld;
    if (col < 4096)      { W = Wq; c = col;        ld = 4096; }
    else if (col < 5120) { W = Wk; c = col - 4096; ld = 1024; }
    else                 { W = Wv; c = col - 5120; ld = 1024; }
    int r0 = chunk * (HID / RCHUNK);
    float4 acc = {0.f, 0.f, 0.f, 0.f};
    for (int i = r0; i < r0 + HID / RCHUNK; ++i) {
        float hv = h[i];
        float4 w = *(const float4*)(W + (size_t)i * ld + c);
        acc.x = fmaf(hv, w.x, acc.x);
        acc.y = fmaf(hv, w.y, acc.y);
        acc.z = fmaf(hv, w.z, acc.z);
        acc.w = fmaf(hv, w.w, acc.w);
    }
    *(float4*)(part + (size_t)chunk * 6144 + col) = acc;
}

__global__ void qkv_reduce_k(const float* __restrict__ part,
                             float* __restrict__ qkv) {
    int col = blockIdx.x * blockDim.x + threadIdx.x;         // 0..6143
    float s = 0.f;
    for (int ch = 0; ch < RCHUNK; ++ch) s += part[(size_t)ch * 6144 + col];
    qkv[col] = s;
}

// ---------------- K2a: RoPE on q and k_new ---------------------------------
__global__ void rope_k(const float* __restrict__ qkv,
                       const float* __restrict__ cosv,
                       const float* __restrict__ sinv,
                       float* __restrict__ qrope,
                       float* __restrict__ krope) {
    for (int i = threadIdx.x; i < NH * HD; i += blockDim.x) {
        int d = i & 127, base = i & ~127;
        float x  = qkv[i];
        float rh = (d < 64) ? -qkv[base + d + 64] : qkv[base + d - 64];
        qrope[i] = x * cosv[d] + rh * sinv[d];
    }
    const float* kn = qkv + 4096;
    for (int i = threadIdx.x; i < NKVH * HD; i += blockDim.x) {
        int d = i & 127, base = i & ~127;
        float x  = kn[i];
        float rh = (d < 64) ? -kn[base + d + 64] : kn[base + d - 64];
        krope[i] = x * cosv[d] + rh * sinv[d];
    }
}

// ---------------- K2b: A^T in bf16 (B-operand staging for WMMA) ------------
__global__ void at_bf16_k(const float* __restrict__ A, __bf16* __restrict__ At) {
    int idx = blockIdx.x * blockDim.x + threadIdx.x;   // 0..65535
    if (idx < RLOW * 1024) {
        int r = idx >> 10, c = idx & 1023;
        At[idx] = (__bf16)A[(size_t)c * RLOW + r];
    }
}

// ---------------- K2c: qg[kvh][r] = sum over 4 rep heads of q_lr -----------
__global__ void qg_k(const float* __restrict__ qrope,
                     const float* __restrict__ A,
                     float* __restrict__ qg) {
    __shared__ float qsum[NKVH * HD];
    for (int i = threadIdx.x; i < NKVH * HD; i += blockDim.x) {
        int kvh = i >> 7, d = i & 127;
        float s = 0.f;
        for (int j = 0; j < 4; ++j) s += qrope[(kvh * 4 + j) * HD + d];
        qsum[i] = s;
    }
    __syncthreads();
    for (int i = threadIdx.x; i < NKVH * RLOW; i += blockDim.x) {
        int kvh = i >> 6, r = i & 63;
        float acc = 0.f;
        for (int d = 0; d < HD; ++d)
            acc = fmaf(qsum[kvh * HD + d], A[(size_t)(kvh * HD + d) * RLOW + r], acc);
        qg[i] = acc;
    }
}

__device__ __forceinline__ void cvt8(const float4 u, const float4 v,
                                     v16bf& a, int off) {
    a[off + 0] = (__bf16)u.x; a[off + 1] = (__bf16)u.y;
    a[off + 2] = (__bf16)u.z; a[off + 3] = (__bf16)u.w;
    a[off + 4] = (__bf16)v.x; a[off + 5] = (__bf16)v.y;
    a[off + 6] = (__bf16)v.z; a[off + 7] = (__bf16)v.w;
}

// ---------------- K3: compressed_k = k_flat @ A  via v_wmma_f32_16x16x32_bf16
// one wave per 16x16 output tile; K = 8 kvh slabs x 128, b128 loads + cvt.
__global__ void compress_wmma_k(const float* __restrict__ kc,
                                const float* __restrict__ krope,
                                const __bf16* __restrict__ At,
                                float* __restrict__ comp) {
    int lane  = threadIdx.x & 31;
    int wid   = blockIdx.x * (blockDim.x >> 5) + (threadIdx.x >> 5);
    int tileN = wid & 3;          // 64/16
    int tileM = wid >> 2;         // 32768/16
    int m0 = tileM << 4, n0 = tileN << 4;
    int g  = lane >> 4;           // half-wave group 0/1
    int mn = lane & 15;
    int row  = m0 + mn;           // A-matrix row (token s)
    int coln = n0 + mn;           // B-matrix column (r)
    v8f acc = {};
    for (int kvh = 0; kvh < NKVH; ++kvh) {
        const float*  abase = (row < SC) ? kc + ((size_t)kvh * SC + row) * HD
                                         : krope + (size_t)kvh * HD;
        const __bf16* bbase = At + (size_t)coln * 1024 + kvh * HD;
#pragma unroll
        for (int kk = 0; kk < HD; kk += 32) {
            int c0 = kk + g * 8;
            int c1 = kk + 16 + g * 8;
            float4 fa0 = *(const float4*)(abase + c0);
            float4 fa1 = *(const float4*)(abase + c0 + 4);
            float4 fa2 = *(const float4*)(abase + c1);
            float4 fa3 = *(const float4*)(abase + c1 + 4);
            v16bf a, b;
            cvt8(fa0, fa1, a, 0);
            cvt8(fa2, fa3, a, 8);
            v8bf b0 = *(const v8bf*)(bbase + c0);
            v8bf b1 = *(const v8bf*)(bbase + c1);
#pragma unroll
            for (int j = 0; j < 8; ++j) { b[j] = b0[j]; b[j + 8] = b1[j]; }
            acc = __builtin_amdgcn_wmma_f32_16x16x32_bf16(
                      false, a, false, b, (short)0, acc, false, false);
        }
    }
#pragma unroll
    for (int i = 0; i < 8; ++i)
        comp[(size_t)(m0 + g * 8 + i) * RLOW + n0 + mn] = acc[i];
}

// ---------------- K4: per-group (GS=4) max of approx scores, all 8 KV heads
__global__ void group_max_k(const float* __restrict__ comp,
                            const float* __restrict__ qg,
                            float* __restrict__ gmax) {
    __shared__ __align__(16) float sqg[NKVH * RLOW];
    for (int i = threadIdx.x; i < NKVH * RLOW; i += blockDim.x) sqg[i] = qg[i];
    __syncthreads();
    int gidx = blockIdx.x * blockDim.x + threadIdx.x;   // 0..8191
    if (gidx >= NGRP) return;
    const float4* sq4 = (const float4*)sqg;
    float best[NKVH];
#pragma unroll
    for (int k = 0; k < NKVH; ++k) best[k] = -INFINITY;
    for (int o = 0; o < 4; ++o) {
        const float4* rowp = (const float4*)(comp + (size_t)(gidx * 4 + o) * RLOW);
        float4 rr[16];
#pragma unroll
        for (int t = 0; t < 16; ++t) rr[t] = rowp[t];
#pragma unroll
        for (int k = 0; k < NKVH; ++k) {
            float dot = 0.f;
#pragma unroll
            for (int t = 0; t < 16; ++t) {
                float4 qv = sq4[k * 16 + t];
                dot = fmaf(rr[t].x, qv.x, dot);
                dot = fmaf(rr[t].y, qv.y, dot);
                dot = fmaf(rr[t].z, qv.z, dot);
                dot = fmaf(rr[t].w, qv.w, dot);
            }
            best[k] = fmaxf(best[k], dot);
        }
    }
#pragma unroll
    for (int k = 0; k < NKVH; ++k) gmax[(size_t)k * NGRP + gidx] = best[k];
}

// ---------------- K5: top-100 groups per KV head (LDS-resident argmax loop)
__global__ void topk_k(const float* __restrict__ gmax, int* __restrict__ tok) {
    int kvh = blockIdx.x, tid = threadIdx.x;
    __shared__ float vals[NGRP];
    __shared__ float rv[256];
    __shared__ int   ri[256];
    for (int i = tid; i < NGRP; i += blockDim.x) vals[i] = gmax[(size_t)kvh * NGRP + i];
    __syncthreads();
    for (int it = 0; it < 100; ++it) {
        float bv = -INFINITY; int bi = 0;
        for (int i = tid; i < NGRP; i += blockDim.x) {
            float v = vals[i];
            if (v > bv) { bv = v; bi = i; }
        }
        rv[tid] = bv; ri[tid] = bi;
        __syncthreads();
        for (int s = 128; s > 0; s >>= 1) {
            if (tid < s) {
                if (rv[tid + s] > rv[tid] ||
                    (rv[tid + s] == rv[tid] && ri[tid + s] < ri[tid])) {
                    rv[tid] = rv[tid + s]; ri[tid] = ri[tid + s];
                }
            }
            __syncthreads();
        }
        if (tid == 0) {
            int gsel = ri[0];
            vals[gsel] = -INFINITY;
            for (int o = 0; o < 4; ++o)
                tok[kvh * NTOK + it * 4 + o] = gsel * 4 + o;
        }
        __syncthreads();
    }
    if (tid < 4) tok[kvh * NTOK + 400 + tid] = STOT - 4 + tid;   // window
}

// ---------------- K6: exact sparse attention, one workgroup per query head
__global__ void sparse_attn_k(const float* __restrict__ qrope,
                              const float* __restrict__ kc,
                              const float* __restrict__ vc,
                              const float* __restrict__ krope,
                              const float* __restrict__ vnew,
                              const int* __restrict__ tok,
                              float* __restrict__ attn) {
    int h = blockIdx.x, kvh = h >> 2, tid = threadIdx.x;
    __shared__ __align__(16) float qs[HD];
    __shared__ float sc[NTOK];
    __shared__ float red[256];
    for (int i = tid; i < HD; i += blockDim.x) qs[i] = qrope[h * HD + i];
    __syncthreads();
    const float scale = 0.08838834764831845f;   // 1/sqrt(128)
    const float4* q4 = (const float4*)qs;
    for (int j = tid; j < NTOK; j += blockDim.x) {
        int t = tok[kvh * NTOK + j];
        const float* krow = (t < SC) ? kc + ((size_t)kvh * SC + t) * HD
                                     : krope + kvh * HD;
        const float4* k4 = (const float4*)krow;
        float d = 0.f;
#pragma unroll
        for (int i = 0; i < HD / 4; ++i) {
            float4 a = q4[i], b = k4[i];
            d = fmaf(a.x, b.x, d);
            d = fmaf(a.y, b.y, d);
            d = fmaf(a.z, b.z, d);
            d = fmaf(a.w, b.w, d);
        }
        sc[j] = d * scale;
    }
    __syncthreads();
    float m = -INFINITY;
    for (int j = tid; j < NTOK; j += blockDim.x) m = fmaxf(m, sc[j]);
    red[tid] = m; __syncthreads();
    for (int s = 128; s > 0; s >>= 1) {
        if (tid < s) red[tid] = fmaxf(red[tid], red[tid + s]);
        __syncthreads();
    }
    float mx = red[0];
    __syncthreads();
    float sum = 0.f;
    for (int j = tid; j < NTOK; j += blockDim.x) {
        float e = __expf(sc[j] - mx);
        sc[j] = e; sum += e;
    }
    red[tid] = sum; __syncthreads();
    for (int s = 128; s > 0; s >>= 1) {
        if (tid < s) red[tid] += red[tid + s];
        __syncthreads();
    }
    float inv = 1.f / red[0];
    __syncthreads();
    if (tid < HD) {
        float acc = 0.f;
        for (int j = 0; j < NTOK; ++j) {
            int t = tok[kvh * NTOK + j];
            const float* vrow = (t < SC) ? vc + ((size_t)kvh * SC + t) * HD
                                         : vnew + kvh * HD;
            acc = fmaf(sc[j], vrow[tid], acc);
        }
        attn[h * HD + tid] = acc * inv;
    }
}

// ---------------- K7: output GEMV against Wo, split-K partials -------------
__global__ void out_gemv_part_k(const float* __restrict__ attn,
                                const float* __restrict__ Wo,
                                float* __restrict__ part) {
    int col = (blockIdx.x * blockDim.x + threadIdx.x) * 4;   // 0..4092
    int chunk = blockIdx.y;                                  // 0..31
    int r0 = chunk * (NH * HD / RCHUNK);
    float4 acc = {0.f, 0.f, 0.f, 0.f};
    for (int i = r0; i < r0 + NH * HD / RCHUNK; ++i) {
        float av = attn[i];
        float4 w = *(const float4*)(Wo + (size_t)i * HID + col);
        acc.x = fmaf(av, w.x, acc.x);
        acc.y = fmaf(av, w.y, acc.y);
        acc.z = fmaf(av, w.z, acc.z);
        acc.w = fmaf(av, w.w, acc.w);
    }
    *(float4*)(part + (size_t)chunk * HID + col) = acc;
}

__global__ void out_reduce_k(const float* __restrict__ part,
                             float* __restrict__ out) {
    int col = blockIdx.x * blockDim.x + threadIdx.x;         // 0..4095
    float s = 0.f;
    for (int ch = 0; ch < RCHUNK; ++ch) s += part[(size_t)ch * HID + col];
    out[col] = s;
}

extern "C" void kernel_launch(void* const* d_in, const int* in_sizes, int n_in,
                              void* d_out, int out_size, void* d_ws, size_t ws_size,
                              hipStream_t stream) {
    const float* hid  = (const float*)d_in[0];
    const float* cosv = (const float*)d_in[1];
    const float* sinv = (const float*)d_in[2];
    const float* kc   = (const float*)d_in[3];
    const float* vc   = (const float*)d_in[4];
    const float* Wq   = (const float*)d_in[5];
    const float* Wk   = (const float*)d_in[6];
    const float* Wv   = (const float*)d_in[7];
    const float* Wo   = (const float*)d_in[8];
    const float* A    = (const float*)d_in[9];
    float* out = (float*)d_out;

    float* ws    = (float*)d_ws;
    float* qkv   = ws;                 // 6144 (q 4096 | k 1024 | v 1024)
    float* qrope = ws + 6144;          // 4096
    float* krope = ws + 10240;         // 1024
    float* qg    = ws + 11264;         // 512
    float* attn  = ws + 11776;         // 4096
    __bf16* At   = (__bf16*)(ws + 16384);   // 64*1024 bf16 (32768 floats)
    float* comp  = ws + 49152;         // 32768*64
    float* gmax  = ws + 2146304;       // 8*8192
    int*   tok   = (int*)(ws + 2211840);    // 8*404 ints
    float* pqkv  = ws + 2215936;       // 32*6144 partials
    float* pout  = ws + 2412544;       // 32*4096 partials
    float* vnew  = qkv + 5120;

    qkv_gemv_part_k<<<dim3(6, RCHUNK),  256, 0, stream>>>(hid, Wq, Wk, Wv, pqkv);
    qkv_reduce_k   <<<24,               256, 0, stream>>>(pqkv, qkv);
    rope_k         <<<1,                256, 0, stream>>>(qkv, cosv, sinv, qrope, krope);
    at_bf16_k      <<<256,              256, 0, stream>>>(A, At);
    qg_k           <<<1,                512, 0, stream>>>(qrope, A, qg);
    compress_wmma_k<<<1024,             256, 0, stream>>>(kc, krope, At, comp);
    group_max_k    <<<32,               256, 0, stream>>>(comp, qg, gmax);
    topk_k         <<<8,                256, 0, stream>>>(gmax, tok);
    sparse_attn_k  <<<32,               256, 0, stream>>>(qrope, kc, vc, krope, vnew, tok, attn);
    out_gemv_part_k<<<dim3(4, RCHUNK),  256, 0, stream>>>(attn, Wo, pout);
    out_reduce_k   <<<16,               256, 0, stream>>>(pout, out);
}
